// N2PAttention_48713519071468
// MI455X (gfx1250) — matrix operand order repaired and checked by reading, not compiled, not run on hardware
//
#include <hip/hip_runtime.h>
#include <cstdint>

#define DEV __device__ __forceinline__

constexpr int Bz = 8, Cz = 128, Nz = 4096, Kz = 32, Hz = 4, Dz = 32;
constexpr int C4 = 4 * Cz;              // 512
constexpr float EPSF = 1e-5f;

typedef unsigned short u16;
typedef __attribute__((ext_vector_type(16))) unsigned short v16u;
typedef __attribute__((ext_vector_type(16))) __bf16        v16bf;
typedef __attribute__((ext_vector_type(8)))  float         v8f;
typedef __attribute__((ext_vector_type(4)))  int           v4i;

// ---- optional CDNA5 async global->LDS path (guarded; fallback keeps compiling) --
#if defined(__has_builtin)
#if __has_builtin(__builtin_amdgcn_global_load_async_to_lds_b128) && \
    __has_builtin(__builtin_amdgcn_s_wait_asynccnt)
#define HAVE_ASYNC_LDS 1
#endif
#endif

#define AS1 __attribute__((address_space(1)))
#define AS3 __attribute__((address_space(3)))
#define TO_GLB(p) ((AS1 v4i*)(uintptr_t)(p))
#define TO_LDS(p) ((AS3 v4i*)(unsigned)(uintptr_t)(p))

// ---- bf16 <-> f32 (round-to-nearest-even), u16 storage to avoid __bf16 arith ----
DEV u16 f2bf(float f) {
  unsigned u = __builtin_bit_cast(unsigned, f);
  u += 0x7FFFu + ((u >> 16) & 1u);
  return (u16)(u >> 16);
}
DEV float bf2f(u16 h) {
  unsigned u = ((unsigned)h) << 16;
  return __builtin_bit_cast(float, u);
}

DEV v8f vzero8() {
  v8f z;
  #pragma unroll
  for (int i = 0; i < 8; ++i) z[i] = 0.f;
  return z;
}

DEV v8f wmma_bf16(v16u a, v16u b, v8f c) {
  return __builtin_amdgcn_wmma_f32_16x16x32_bf16(
      false, __builtin_bit_cast(v16bf, a),
      false, __builtin_bit_cast(v16bf, b),
      (short)0, c, false, false);
}

// ---- WMMA fragment loaders (contiguous k = 16*half + s in A and B) -------------------------------
// A fragment 16x32 bf16 from LDS tile laid out [row][ld] (row-major, K contiguous)
DEV v16u load_a_frag(const u16* base, int ld, int mrow, int kcol) {
  int lane = threadIdx.x & 31;
  int half = lane >> 4;            // lanes 16-31 hold K+16..K+31
  int m    = lane & 15;
  const u16* rp = base + (size_t)(mrow + m) * ld + kcol;
  v16u a;
  #pragma unroll
  for (int v = 0; v < 8; ++v) {
    int k = 2 * v + 16 * half;
    a[2 * v]     = rp[k];
    a[2 * v + 1] = rp[k + 1];
  }
  return a;
}
// B fragment 32x16 from LDS tile stored [n][ld] (i.e. transposed: B[k][n]=tile[n][k])
DEV v16u load_b_frag_t(const u16* base, int ld, int kcol, int ncol) {
  int lane = threadIdx.x & 31;
  int n  = ncol + (lane & 15);
  int kb = kcol + ((lane >> 4) << 4);
  const u16* rp = base + (size_t)n * ld + kb;
  v16u b;
  #pragma unroll
  for (int j = 0; j < 16; ++j) b[j] = rp[j];
  return b;
}
// B fragment 32x16 from tile stored [k][ld] (row-major K x N)
DEV v16u load_b_frag(const u16* base, int ld, int kcol, int ncol) {
  int lane = threadIdx.x & 31;
  int n  = ncol + (lane & 15);
  int kb = kcol + ((lane >> 4) << 4);
  v16u b;
  #pragma unroll
  for (int j = 0; j < 16; ++j) b[j] = base[(size_t)(kb + j) * ld + n];
  return b;
}

// ---- weight convert + transpose: WT[c][o] = (bf16)W[o][c] ----------------------
__global__ void k_cvt_wT(const float* __restrict__ W, u16* __restrict__ WT,
                         int O, int Cc) {
  int i = blockIdx.x * blockDim.x + threadIdx.x;
  if (i >= O * Cc) return;
  int o = i / Cc, c = i % Cc;
  WT[(size_t)c * O + o] = f2bf(W[i]);
}

// ---- x (B,C,N) -> pcd bf16 (B,N,C) + squared norms -----------------------------
__global__ __launch_bounds__(256)
void k_prep(const float* __restrict__ x, u16* __restrict__ pcdb,
            float* __restrict__ sq) {
  __shared__ float red[32][8];
  int b = blockIdx.y;
  int n = blockIdx.x * 32 + (threadIdx.x >> 3);
  int l8 = threadIdx.x & 7;
  const float* xb = x + (size_t)b * Cz * Nz + n;
  u16* pr = pcdb + ((size_t)b * Nz + n) * Cz;
  float s = 0.f;
  #pragma unroll
  for (int j = 0; j < 16; ++j) {
    int c = l8 * 16 + j;
    float v = xb[(size_t)c * Nz];
    pr[c] = f2bf(v);
    s += v * v;
  }
  red[threadIdx.x >> 3][l8] = s;
  __syncthreads();
  if (l8 == 0) {
    float t = 0.f;
    #pragma unroll
    for (int j = 0; j < 8; ++j) t += red[threadIdx.x >> 3][j];
    sq[(size_t)b * Nz + n] = t;
  }
}

// ---- generic WMMA GEMM:  out(M x NCOLS) = A(M x KDIM) * Bw(KDIM x NCOLS) -------
// EPI 0: store bf16   EPI 1: leaky-relu(0.2) + bf16   EPI 2: +resid, f32 (B,C,N)
template <int KDIM, int NCOLS, int EPI>
__global__ __launch_bounds__(256)
void k_gemm(const u16* __restrict__ A, const u16* __restrict__ Bw,
            u16* __restrict__ outB, const float* __restrict__ resid,
            float* __restrict__ outF) {
  __shared__ u16 As[128 * 128];
  __shared__ u16 Bs[128 * 128];
  int m0  = blockIdx.x * 128;
  int tid = threadIdx.x;
  int wid = tid >> 5, lane = tid & 31;
  for (int nc = 0; nc < NCOLS; nc += 128) {
    v8f acc[8];
    #pragma unroll
    for (int t = 0; t < 8; ++t) acc[t] = vzero8();
    for (int kc = 0; kc < KDIM; kc += 128) {
      __syncthreads();
#if HAVE_ASYNC_LDS
      #pragma unroll
      for (int e = tid * 8; e < 128 * 128; e += 256 * 8) {
        int r = e >> 7, c = e & 127;
        __builtin_amdgcn_global_load_async_to_lds_b128(
            TO_GLB(&A[(size_t)(m0 + r) * KDIM + kc + c]), TO_LDS(&As[e]), 0, 0);
        __builtin_amdgcn_global_load_async_to_lds_b128(
            TO_GLB(&Bw[(size_t)(kc + r) * NCOLS + nc + c]), TO_LDS(&Bs[e]), 0, 0);
      }
      __builtin_amdgcn_s_wait_asynccnt(0);
#else
      #pragma unroll
      for (int e = tid * 8; e < 128 * 128; e += 256 * 8) {
        int r = e >> 7, c = e & 127;
        *(uint4*)&As[e] = *(const uint4*)&A[(size_t)(m0 + r) * KDIM + kc + c];
        *(uint4*)&Bs[e] = *(const uint4*)&Bw[(size_t)(kc + r) * NCOLS + nc + c];
      }
#endif
      __syncthreads();
      #pragma unroll
      for (int kk = 0; kk < 128; kk += 32) {
        v16u af = load_a_frag(As, 128, wid * 16, kk);
        #pragma unroll
        for (int nt = 0; nt < 8; ++nt) {
          v16u bfr = load_b_frag(Bs, 128, kk, nt * 16);
          acc[nt] = wmma_bf16(af, bfr, acc[nt]);
        }
      }
    }
    int colb = lane & 15, rb = (lane >> 4) * 8;
    #pragma unroll
    for (int nt = 0; nt < 8; ++nt) {
      #pragma unroll
      for (int r = 0; r < 8; ++r) {
        int m   = m0 + wid * 16 + rb + r;
        int col = nc + nt * 16 + colb;
        float v = acc[nt][r];
        if constexpr (EPI == 0) {
          outB[(size_t)m * NCOLS + col] = f2bf(v);
        } else if constexpr (EPI == 1) {
          v = v > 0.f ? v : 0.2f * v;
          outB[(size_t)m * NCOLS + col] = f2bf(v);
        } else {
          int b = m / Nz, n = m % Nz;
          size_t gi = ((size_t)b * Cz + col) * Nz + n;
          outF[gi] = resid[gi] + v;
        }
      }
    }
  }
}

// ---- fused gram-matrix + top-K neighbor selection ------------------------------
__global__ __launch_bounds__(256)
void k_dist_topk(const u16* __restrict__ pcdb, const float* __restrict__ sq,
                 int* __restrict__ idxout) {
  __shared__ u16   Aq[16 * 128];
  __shared__ u16   Bc[128 * 128];
  __shared__ float sc[16 * 128];
  __shared__ float sqq[16];
  __shared__ float sqc[128];
  __shared__ float topv[16][32];
  __shared__ int   topi[16][32];
  int b = blockIdx.y, q0 = blockIdx.x * 16, tid = threadIdx.x;
  int wid = tid >> 5, lane = tid & 31;
  const u16* P = pcdb + (size_t)b * Nz * Cz;
  { int e = tid * 8;
    if (e < 16 * 128) {
      int r = e >> 7, c = e & 127;
      *(uint4*)&Aq[e] = *(const uint4*)&P[(size_t)(q0 + r) * Cz + c];
    } }
  if (tid < 16) sqq[tid] = sq[(size_t)b * Nz + q0 + tid];
  for (int e = tid; e < 512; e += 256) {
    topv[e >> 5][e & 31] = -3.4e38f;
    topi[e >> 5][e & 31] = 0;
  }
  for (int ch = 0; ch < Nz; ch += 128) {
    __syncthreads();
#if HAVE_ASYNC_LDS
    #pragma unroll
    for (int e = tid * 8; e < 128 * 128; e += 2048) {
      int r = e >> 7, c = e & 127;
      __builtin_amdgcn_global_load_async_to_lds_b128(
          TO_GLB(&P[(size_t)(ch + r) * Cz + c]), TO_LDS(&Bc[e]), 0, 0);
    }
#else
    #pragma unroll
    for (int e = tid * 8; e < 128 * 128; e += 2048) {
      int r = e >> 7, c = e & 127;
      *(uint4*)&Bc[e] = *(const uint4*)&P[(size_t)(ch + r) * Cz + c];
    }
#endif
    if (tid < 128) sqc[tid] = sq[(size_t)b * Nz + ch + tid];
    if (ch + 128 < Nz)
      __builtin_prefetch(&P[(size_t)(ch + 128) * Cz + tid * 8], 0, 1);
#if HAVE_ASYNC_LDS
    __builtin_amdgcn_s_wait_asynccnt(0);
#endif
    __syncthreads();
    v8f acc = vzero8();
    #pragma unroll
    for (int kk = 0; kk < 128; kk += 32) {
      v16u af  = load_a_frag(Aq, 128, 0, kk);
      v16u bfr = load_b_frag_t(Bc, 128, kk, wid * 16);
      acc = wmma_bf16(af, bfr, acc);
    }
    int coln = wid * 16 + (lane & 15);
    int rb   = (lane >> 4) * 8;
    #pragma unroll
    for (int r = 0; r < 8; ++r) {
      int row = rb + r;
      sc[row * 128 + coln] = 2.f * acc[r] - sqq[row] - sqc[coln];
    }
    __syncthreads();
    if (tid < 16) {                    // per-row insertion into sorted top-32
      int row = tid;
      for (int j = 0; j < 128; ++j) {
        float v = sc[row * 128 + j];
        if (v > topv[row][31]) {
          int pos = 31;
          while (pos > 0 && topv[row][pos - 1] < v) {
            topv[row][pos] = topv[row][pos - 1];
            topi[row][pos] = topi[row][pos - 1];
            --pos;
          }
          topv[row][pos] = v;
          topi[row][pos] = ch + j;
        }
      }
    }
  }
  __syncthreads();
  for (int e = tid; e < 512; e += 256) {
    int row = e >> 5, kk = e & 31;
    idxout[((size_t)b * Nz + q0 + row) * Kz + kk] = topi[row][kk];
  }
}

// ---- fused gather-diff + K/V WMMA projections + attention ----------------------
__global__ __launch_bounds__(128)
void k_attn(const u16* __restrict__ pcdb, const u16* __restrict__ qb,
            const int* __restrict__ idxin, const u16* __restrict__ WkT,
            const u16* __restrict__ WvT, float* __restrict__ attnout) {
  __shared__ u16   dl[32 * 128];          // neighbor diffs, bf16
  __shared__ float kf[Hz][32][32];
  __shared__ float vf[Hz][32][32];
  __shared__ float qh[Cz];
  __shared__ float att[Hz][32];
  __shared__ int   nbi[32];
  int b = blockIdx.y, n = blockIdx.x, tid = threadIdx.x;
  int h = tid >> 5, lane = tid & 31;
  const u16* P = pcdb + (size_t)b * Nz * Cz;
  if (tid < 32)  nbi[tid] = idxin[((size_t)b * Nz + n) * Kz + tid];
  if (tid < 128) qh[tid]  = bf2f(qb[((size_t)b * Nz + n) * Cz + tid]);
  __syncthreads();
  const u16* cen = P + (size_t)n * Cz;
  for (int e = tid; e < 32 * 128; e += 128) {
    int r = e >> 7, c = e & 127;
    dl[e] = f2bf(bf2f(P[(size_t)nbi[r] * Cz + c]) - bf2f(cen[c]));
  }
  __syncthreads();
  // per-head wave: kf/vf = diff(32x128) @ W^T[:, h*32 : h*32+32]
  v8f ak[2][2], av[2][2];
  #pragma unroll
  for (int mt = 0; mt < 2; ++mt)
    for (int dt = 0; dt < 2; ++dt) { ak[mt][dt] = vzero8(); av[mt][dt] = vzero8(); }
  #pragma unroll
  for (int kk = 0; kk < 128; kk += 32) {
    v16u a0 = load_a_frag(dl, 128, 0, kk);
    v16u a1 = load_a_frag(dl, 128, 16, kk);
    #pragma unroll
    for (int dt = 0; dt < 2; ++dt) {
      v16u bk = load_b_frag(WkT, Cz, kk, h * 32 + dt * 16);
      ak[0][dt] = wmma_bf16(a0, bk, ak[0][dt]);
      ak[1][dt] = wmma_bf16(a1, bk, ak[1][dt]);
      v16u bv = load_b_frag(WvT, Cz, kk, h * 32 + dt * 16);
      av[0][dt] = wmma_bf16(a0, bv, av[0][dt]);
      av[1][dt] = wmma_bf16(a1, bv, av[1][dt]);
    }
  }
  int colb = lane & 15, rb = (lane >> 4) * 8;
  #pragma unroll
  for (int mt = 0; mt < 2; ++mt) {
    #pragma unroll
    for (int dt = 0; dt < 2; ++dt) {
      #pragma unroll
      for (int r = 0; r < 8; ++r) {
        kf[h][mt * 16 + rb + r][dt * 16 + colb] = ak[mt][dt][r];
        vf[h][mt * 16 + rb + r][dt * 16 + colb] = av[mt][dt][r];
      }
    }
  }
  __syncthreads();
  // softmax over K=32 (one lane per neighbor, wave32 reductions)
  float e = 0.f;
  #pragma unroll
  for (int d = 0; d < Dz; ++d) e += qh[h * 32 + d] * kf[h][lane][d];
  e *= 0.17677669529663687f;                      // 1/sqrt(D)
  float mx = e;
  for (int off = 16; off > 0; off >>= 1) mx = fmaxf(mx, __shfl_xor(mx, off, 32));
  float p = __expf(e - mx);
  float s = p;
  for (int off = 16; off > 0; off >>= 1) s += __shfl_xor(s, off, 32);
  att[h][lane] = p / s;
  __syncthreads();
  float o = 0.f;
  #pragma unroll
  for (int kI = 0; kI < Kz; ++kI) o += att[h][kI] * vf[h][kI][lane];
  attnout[((size_t)b * Cz + (h * 32 + lane)) * Nz + n] = o;
}

// ---- batch-norm statistics: one block per channel ------------------------------
__global__ __launch_bounds__(256)
void k_bn_stats(const float* __restrict__ y, const float* __restrict__ yadd,
                int has_add, float* __restrict__ meanv, float* __restrict__ varv) {
  __shared__ float s1[256], s2[256];
  int c = blockIdx.x, tid = threadIdx.x;
  float a1 = 0.f, a2 = 0.f;
  for (int b = 0; b < Bz; ++b) {
    const float* p1 = y + ((size_t)b * Cz + c) * Nz;
    const float* p2 = yadd ? yadd + ((size_t)b * Cz + c) * Nz : nullptr;
    for (int i = tid; i < Nz; i += 256) {
      float v = p1[i];
      if (has_add) v += p2[i];
      a1 += v; a2 += v * v;
    }
  }
  s1[tid] = a1; s2[tid] = a2;
  __syncthreads();
  for (int st = 128; st > 0; st >>= 1) {
    if (tid < st) { s1[tid] += s1[tid + st]; s2[tid] += s2[tid + st]; }
    __syncthreads();
  }
  if (tid == 0) {
    float m = s1[0] / (float)(Bz * Nz);
    meanv[c] = m;
    varv[c]  = s2[0] / (float)(Bz * Nz) - m * m;
  }
}

__global__ __launch_bounds__(256)
void k_bn1_apply(const float* __restrict__ x, const float* __restrict__ attnout,
                 const float* __restrict__ meanv, const float* __restrict__ varv,
                 const float* __restrict__ g, const float* __restrict__ bb,
                 float* __restrict__ x1f, u16* __restrict__ x1b) {
  size_t i = (size_t)blockIdx.x * 256 + threadIdx.x;
  size_t n = i % Nz, c = (i / Nz) % Cz, b = i / ((size_t)Nz * Cz);
  float v  = x[i] + attnout[i];
  float xn = g[c] * (v - meanv[c]) * rsqrtf(varv[c] + EPSF) + bb[c];
  x1f[i] = xn;
  x1b[((size_t)b * Nz + n) * Cz + c] = f2bf(xn);
}

__global__ __launch_bounds__(256)
void k_bn2_apply(const float* __restrict__ y2, const float* __restrict__ meanv,
                 const float* __restrict__ varv, const float* __restrict__ g,
                 const float* __restrict__ bb, float* __restrict__ out) {
  size_t i = (size_t)blockIdx.x * 256 + threadIdx.x;
  size_t c = (i / Nz) % Cz;
  out[i] = g[c] * (y2[i] - meanv[c]) * rsqrtf(varv[c] + EPSF) + bb[c];
}

// ================================================================================
extern "C" void kernel_launch(void* const* d_in, const int* in_sizes, int n_in,
                              void* d_out, int out_size, void* d_ws, size_t ws_size,
                              hipStream_t stream) {
  (void)in_sizes; (void)n_in; (void)out_size; (void)ws_size;
  const float* x  = (const float*)d_in[0];
  const float* Wq = (const float*)d_in[1];
  const float* Wk = (const float*)d_in[2];
  const float* Wv = (const float*)d_in[3];
  const float* W1 = (const float*)d_in[4];
  const float* W2 = (const float*)d_in[5];
  const float* g1 = (const float*)d_in[6];
  const float* b1 = (const float*)d_in[7];
  const float* g2 = (const float*)d_in[8];
  const float* b2 = (const float*)d_in[9];
  float* out = (float*)d_out;

  char* ws = (char*)d_ws;
  size_t off = 0;
  auto take = [&](size_t bytes) -> char* {
    char* p = ws + off;
    off = (off + bytes + 255) & ~(size_t)255;
    return p;
  };
  u16*   WQT     = (u16*)take((size_t)Cz * Cz * 2);
  u16*   WKT     = (u16*)take((size_t)Cz * Cz * 2);
  u16*   WVT     = (u16*)take((size_t)Cz * Cz * 2);
  u16*   W1T     = (u16*)take((size_t)Cz * C4 * 2);
  u16*   W2T     = (u16*)take((size_t)C4 * Cz * 2);
  u16*   pcdb    = (u16*)take((size_t)Bz * Nz * Cz * 2);
  float* sq      = (float*)take((size_t)Bz * Nz * 4);
  u16*   qb      = (u16*)take((size_t)Bz * Nz * Cz * 2);
  int*   idx     = (int*)take((size_t)Bz * Nz * Kz * 4);
  float* attnout = (float*)take((size_t)Bz * Cz * Nz * 4);
  float* x1f     = (float*)take((size_t)Bz * Cz * Nz * 4);
  u16*   x1b     = (u16*)take((size_t)Bz * Nz * Cz * 2);
  u16*   hb      = (u16*)take((size_t)Bz * Nz * C4 * 2);
  float* y2f     = (float*)take((size_t)Bz * Cz * Nz * 4);
  float* mean1   = (float*)take(Cz * 4);
  float* var1    = (float*)take(Cz * 4);
  float* mean2   = (float*)take(Cz * 4);
  float* var2    = (float*)take(Cz * 4);

  k_cvt_wT<<<64,  256, 0, stream>>>(Wq, WQT, Cz, Cz);
  k_cvt_wT<<<64,  256, 0, stream>>>(Wk, WKT, Cz, Cz);
  k_cvt_wT<<<64,  256, 0, stream>>>(Wv, WVT, Cz, Cz);
  k_cvt_wT<<<256, 256, 0, stream>>>(W1, W1T, C4, Cz);
  k_cvt_wT<<<256, 256, 0, stream>>>(W2, W2T, Cz, C4);

  k_prep<<<dim3(Nz / 32, Bz), 256, 0, stream>>>(x, pcdb, sq);

  k_gemm<Cz, Cz, 0><<<(Bz * Nz) / 128, 256, 0, stream>>>(pcdb, WQT, qb, nullptr, nullptr);

  k_dist_topk<<<dim3(Nz / 16, Bz), 256, 0, stream>>>(pcdb, sq, idx);

  k_attn<<<dim3(Nz, Bz), 128, 0, stream>>>(pcdb, qb, idx, WKT, WVT, attnout);

  k_bn_stats<<<Cz, 256, 0, stream>>>(x, attnout, 1, mean1, var1);
  k_bn1_apply<<<(Bz * Cz * Nz) / 256, 256, 0, stream>>>(x, attnout, mean1, var1,
                                                        g1, b1, x1f, x1b);

  k_gemm<Cz, C4, 1><<<(Bz * Nz) / 128, 256, 0, stream>>>(x1b, W1T, hb, nullptr, nullptr);
  k_gemm<C4, Cz, 2><<<(Bz * Nz) / 128, 256, 0, stream>>>(hb, W2T, nullptr, x1f, y2f);

  k_bn_stats<<<Cz, 256, 0, stream>>>(y2f, nullptr, 0, mean2, var2);
  k_bn2_apply<<<(Bz * Cz * Nz) / 256, 256, 0, stream>>>(y2f, mean2, var2, g2, b2, out);
}
